// RelativeMultiHeadAttention_12343736008788
// MI455X (gfx1250) — matrix operand order repaired and checked by reading, not compiled, not run on hardware
//
#include <hip/hip_runtime.h>
#include <hip/hip_bf16.h>

// ---------------------------------------------------------------------------
// RelativeMultiHeadAttention for MI455X (gfx1250), bf16 WMMA pipeline.
// B=4, S=1024, E=1024, H=16, HD=64, rel_table = (2047, 64) (full table used).
//
// Roofline: ~60 GFLOP total, ~100 MB unique HBM traffic (everything fits the
// 192 MB L2) -> compute-bound on WMMA. bf16 16x16x32 WMMA with f32 accum,
// flash softmax (no 64x1024x1024 score tensor), rel-shift handled as a
// per-q-tile 16x1040 positional window in LDS.
// ---------------------------------------------------------------------------

#define B_ 4
#define S_ 1024
#define E_ 1024
#define H_ 16
#define HD_ 64
#define M_ (B_ * S_)          // 4096 rows in (b,s) space
#define PROWS 2047            // rel table rows
#define PWIN 1040             // per-q-tile positional window (15 + 1024 + pad)

typedef __attribute__((ext_vector_type(16))) __bf16 bf16x16;
typedef __attribute__((ext_vector_type(8)))  __bf16 bf16x8;
typedef __attribute__((ext_vector_type(2)))  __bf16 bf16x2;
typedef __attribute__((ext_vector_type(8)))  float  f32x8;

// ---- bf16 helpers (round-to-nearest-even) ---------------------------------
// Prefer the hardware packed converter (v_cvt_pk_bf16_f32) when declared.
#if __has_builtin(__builtin_amdgcn_cvt_pk_bf16_f32)
#define HAVE_HW_BF16_CVT 1
#else
#define HAVE_HW_BF16_CVT 0
#endif

static __device__ __forceinline__ __bf16 f2bf(float f) {
#if HAVE_HW_BF16_CVT
  auto r = __builtin_amdgcn_cvt_pk_bf16_f32(f, 0.f);
  return r[0];
#else
  union { float f; unsigned u; } x; x.f = f;
  unsigned r = (x.u + 0x7FFFu + ((x.u >> 16) & 1u)) >> 16;
  union { unsigned short s; __bf16 b; } y; y.s = (unsigned short)r;
  return y.b;
#endif
}
static __device__ __forceinline__ bf16x2 f2bf2(float a, float b) {
#if HAVE_HW_BF16_CVT
  auto r = __builtin_amdgcn_cvt_pk_bf16_f32(a, b);
  bf16x2 o; o[0] = r[0]; o[1] = r[1];
  return o;
#else
  bf16x2 o; o[0] = f2bf(a); o[1] = f2bf(b);
  return o;
#endif
}
static __device__ __forceinline__ float bf2f(__bf16 b) {
  union { unsigned short s; __bf16 b; } y; y.b = b;
  union { unsigned u; float f; } x; x.u = ((unsigned)y.s) << 16;
  return x.f;
}

static __device__ __forceinline__ bf16x16 cat8(bf16x8 lo, bf16x8 hi) {
  return __builtin_shufflevector(lo, hi, 0,1,2,3,4,5,6,7,8,9,10,11,12,13,14,15);
}

// ---- WMMA fragment loaders (CDNA5 ISA §7.12.2, 16-bit) ---------
// A 16x32 (MxK): lanes 0-15 -> M=lane, K={k0..k0+7, k0+16..k0+23};
//                lanes 16-31 -> M=lane-16, K={k0+8..k0+15, k0+24..k0+31}.
static __device__ __forceinline__ bf16x16
load_a_frag(const __bf16* base, int ld, int k0, int lane) {
  int m  = lane & 15;
  int kb = k0 + ((lane & 16) ? 8 : 0);
  const __bf16* p = base + m * ld + kb;
  return cat8(*(const bf16x8*)p, *(const bf16x8*)(p + 16));
}
// B 32x16 (KxN): lane holds column n=lane&15; lanes 0-15 K={k0..k0+7, k0+16..k0+23},
// lanes 16-31 K={k0+8..k0+15, k0+24..k0+31} (same K order as the A fragment).
// `row` must already point at the memory row holding column n (i.e. W[n][*]).
static __device__ __forceinline__ bf16x16
load_b_row(const __bf16* row, int k0, int lane) {
  const __bf16* p = row + k0 + ((lane & 16) ? 8 : 0);
  return cat8(*(const bf16x8*)p, *(const bf16x8*)(p + 16));
}

static __device__ __forceinline__ f32x8
wmma_bf16(bf16x16 a, bf16x16 b, f32x8 c) {
  // (neg_a, A, neg_b, B, c_mod, C, reuse_a, reuse_b)
  return __builtin_amdgcn_wmma_f32_16x16x32_bf16(false, a, false, b,
                                                 (short)0, c, false, false);
}

// ---------------------------------------------------------------------------
// Kernel 1: fp32 -> bf16 convert, 2 elements/thread via packed cvt + b32 store
// (all sizes here are multiples of 2)
// ---------------------------------------------------------------------------
__global__ void cvt_f32_bf16(const float* __restrict__ in,
                             __bf16* __restrict__ out, int n2) {
  int i = blockIdx.x * blockDim.x + threadIdx.x;
  if (i < n2) {
    const float2 v = ((const float2*)in)[i];
    ((bf16x2*)out)[i] = f2bf2(v.x, v.y);
  }
}

// ---------------------------------------------------------------------------
// Kernel 2: rel_key = rel_table @ Wp^T  (2047 x 64) -> bf16
// ---------------------------------------------------------------------------
__global__ void relkey_kernel(const float* __restrict__ rel,
                              const float* __restrict__ Wp,
                              __bf16* __restrict__ out) {
  int i = blockIdx.x * blockDim.x + threadIdx.x;
  if (i >= PROWS * HD_) return;
  int p = i >> 6, d = i & 63;
  float s = 0.f;
  #pragma unroll 8
  for (int k = 0; k < HD_; ++k) s += rel[p * HD_ + k] * Wp[d * HD_ + k];
  out[i] = f2bf(s);
}

// ---------------------------------------------------------------------------
// Kernel 3: GEMM  C[m,n] = (sum_k A[m,k]*W[n,k] + bias[n]) * scale
// A: (4096 x 1024) bf16 row-major, W: (1024 x 1024) bf16 row-major.
// 4 waves/block; each wave owns a 32x64 tile: 2 A-frags x 4 B-frags per
// k-step -> 8 WMMA per 12 b128 loads.
// MODE 0: store bf16 as (b,h,s,d)   [Q (pre-scaled) and K]
// MODE 2: store bf16 as (b,h,d,s)   [V transposed, makes probs@V K-contig]
// MODE 3: store fp32 row-major (b,s,e) [final output]
// ---------------------------------------------------------------------------
template <int MODE>
__global__ __launch_bounds__(128) void gemm_wt(const __bf16* __restrict__ A,
                                               const __bf16* __restrict__ W,
                                               const float* __restrict__ bias,
                                               void* __restrict__ out,
                                               float scale) {
  const int lane = threadIdx.x & 31;
  const int wave = threadIdx.x >> 5;
  const int m0 = blockIdx.x * 128 + wave * 32;   // 32 rows per wave
  const int n0 = blockIdx.y * 64;                // 64 cols per block
  const int lc = lane & 15, half = (lane >> 4) & 1;

  const __bf16* Ab0 = A + (long)m0 * E_;
  const __bf16* Ab1 = Ab0 + (long)16 * E_;
  const __bf16* wr[4];
  #pragma unroll
  for (int j = 0; j < 4; ++j) wr[j] = W + (long)(n0 + j * 16 + lc) * E_;

  f32x8 acc[2][4];
  #pragma unroll
  for (int t = 0; t < 2; ++t)
    #pragma unroll
    for (int j = 0; j < 4; ++j) acc[t][j] = f32x8{};

  #pragma unroll 2
  for (int k = 0; k < E_; k += 32) {
    const bf16x16 a0 = load_a_frag(Ab0, E_, k, lane);
    const bf16x16 a1 = load_a_frag(Ab1, E_, k, lane);
    #pragma unroll
    for (int j = 0; j < 4; ++j) {
      const bf16x16 b = load_b_row(wr[j], k, lane);
      acc[0][j] = wmma_bf16(a0, b, acc[0][j]);
      acc[1][j] = wmma_bf16(a1, b, acc[1][j]);
    }
    if (k + 256 < E_) {  // L2 prefetch hint -> global_prefetch_b8
      __builtin_prefetch(Ab0 + lc * E_ + k + 256, 0, 1);
      __builtin_prefetch(Ab1 + lc * E_ + k + 256, 0, 1);
    }
  }

  #pragma unroll
  for (int t = 0; t < 2; ++t) {
    #pragma unroll
    for (int j = 0; j < 4; ++j) {
      const int n = n0 + j * 16 + lc;
      const float bv = bias[n];
      #pragma unroll
      for (int v = 0; v < 8; ++v) {
        const int m = m0 + t * 16 + v + 8 * half;
        const float val = (acc[t][j][v] + bv) * scale;
        if (MODE == 3) {
          ((float*)out)[(long)m * E_ + n] = val;
        } else {
          const int b = m >> 10, s = m & 1023, h = n >> 6, d = n & 63;
          __bf16* o = (__bf16*)out;
          if (MODE == 2)
            o[(((long)b * H_ + h) * HD_ + d) * S_ + s] = f2bf(val);
          else
            o[(((long)b * H_ + h) * S_ + s) * HD_ + d] = f2bf(val);
        }
      }
    }
  }
}

// ---------------------------------------------------------------------------
// Kernel 4: fused relative attention, one wave per (b,h,q-tile).
//  - pos window PW[qi][j] = Qs[qi] . relk[pbase+j]  (Qs pre-scaled by 1/8)
//    into LDS; shifted gather j = 15 - qi + k  (== p = S-1-q+k).
//  - flash softmax over 1024 keys in chunks of 32 (2 score frags).
//  - probs transposed through LDS into an A-fragment, ctx += P @ V.
// ---------------------------------------------------------------------------
__global__ __launch_bounds__(32) void attn_kernel(const __bf16* __restrict__ Q,
                                                  const __bf16* __restrict__ K,
                                                  const __bf16* __restrict__ Vt,
                                                  const __bf16* __restrict__ relk,
                                                  __bf16* __restrict__ ctx) {
  __shared__ __align__(16) __bf16 PW[16 * PWIN];  // 33280 B pos window
  __shared__ __align__(16) __bf16 Pb[16 * 32];    // 1 KB probs transpose buf

  const int lane = threadIdx.x;
  const int lc = lane & 15, half = (lane >> 4) & 1;
  const int q0 = blockIdx.x * 16;
  const int bh = blockIdx.y;           // b*16 + h
  const int b = bh >> 4, h = bh & 15;

  const __bf16* Qb = Q + ((long)bh * S_ + q0) * HD_;
  const __bf16* Kb = K + (long)bh * S_ * HD_;
  const __bf16* Vb = Vt + (long)bh * HD_ * S_;

  // Q tile as two A fragments (d = 0..31, 32..63); Q already scaled by 1/8.
  const bf16x16 aq0 = load_a_frag(Qb, HD_, 0, lane);
  const bf16x16 aq1 = load_a_frag(Qb, HD_, 32, lane);

  // ---- positional window: PW[m][j], p = pbase + j, pbase = S-16-q0 --------
  const int pbase = S_ - 16 - q0;
  for (int nt = 0; nt < PWIN / 16; ++nt) {
    int prow = pbase + nt * 16 + lc;
    prow = prow < 0 ? 0 : (prow > PROWS - 1 ? PROWS - 1 : prow);  // pad rows
    const __bf16* row = relk + (long)prow * HD_;
    f32x8 c = {};
    c = wmma_bf16(aq0, load_b_row(row, 0, lane), c);
    c = wmma_bf16(aq1, load_b_row(row, 32, lane), c);
    #pragma unroll
    for (int v = 0; v < 8; ++v)
      PW[(v + 8 * half) * PWIN + nt * 16 + lc] = f2bf(c[v]);
  }
  __syncthreads();

  // ---- flash softmax over all 1024 keys, 32 per iteration -----------------
  f32x8 acc[4] = {f32x8{}, f32x8{}, f32x8{}, f32x8{}};
  float rm[8], rs[8];
  #pragma unroll
  for (int v = 0; v < 8; ++v) { rm[v] = -1e30f; rs[v] = 0.f; }

  for (int kt = 0; kt < S_; kt += 32) {
    // content scores (already scaled via Q): two 16x16 fragments
    f32x8 s0 = {}, s1 = {};
    {
      const __bf16* kr0 = Kb + (long)(kt + lc) * HD_;
      const __bf16* kr1 = Kb + (long)(kt + 16 + lc) * HD_;
      s0 = wmma_bf16(aq0, load_b_row(kr0, 0, lane), s0);
      s0 = wmma_bf16(aq1, load_b_row(kr0, 32, lane), s0);
      s1 = wmma_bf16(aq0, load_b_row(kr1, 0, lane), s1);
      s1 = wmma_bf16(aq1, load_b_row(kr1, 32, lane), s1);
    }
    // add rel-shifted positional scores: j = 15 - m + k_local
    #pragma unroll
    for (int v = 0; v < 8; ++v) {
      const int m = v + 8 * half;
      s0[v] += bf2f(PW[m * PWIN + 15 - m + kt + lc]);
      s1[v] += bf2f(PW[m * PWIN + 15 - m + kt + 16 + lc]);
    }
    // online softmax update (row reductions across the 16-lane half)
    float al[8];
    #pragma unroll
    for (int v = 0; v < 8; ++v) {
      float cm = fmaxf(s0[v], s1[v]);
      cm = fmaxf(cm, __shfl_xor(cm, 1, 32));
      cm = fmaxf(cm, __shfl_xor(cm, 2, 32));
      cm = fmaxf(cm, __shfl_xor(cm, 4, 32));
      cm = fmaxf(cm, __shfl_xor(cm, 8, 32));
      const float mn = fmaxf(rm[v], cm);
      al[v] = __expf(rm[v] - mn);
      rm[v] = mn;
      const float p0 = __expf(s0[v] - mn);
      const float p1 = __expf(s1[v] - mn);
      s0[v] = p0; s1[v] = p1;
      float sum = p0 + p1;
      sum += __shfl_xor(sum, 1, 32);
      sum += __shfl_xor(sum, 2, 32);
      sum += __shfl_xor(sum, 4, 32);
      sum += __shfl_xor(sum, 8, 32);
      rs[v] = rs[v] * al[v] + sum;
    }
    #pragma unroll
    for (int db = 0; db < 4; ++db)
      #pragma unroll
      for (int v = 0; v < 8; ++v) acc[db][v] *= al[v];

    // transpose probs (C-layout) -> A-fragment layout via LDS
    __syncthreads();
    #pragma unroll
    for (int v = 0; v < 8; ++v) {
      const int m = v + 8 * half;
      Pb[m * 32 + lc]      = f2bf(s0[v]);
      Pb[m * 32 + 16 + lc] = f2bf(s1[v]);
    }
    __syncthreads();
    const bf16x16 ap = load_a_frag(Pb, 32, 0, lane);

    // ctx += P @ V  (V stored transposed (d, s): K-contiguous B operand)
    #pragma unroll
    for (int db = 0; db < 4; ++db) {
      const __bf16* vr = Vb + (long)(db * 16 + lc) * S_;
      acc[db] = wmma_bf16(ap, load_b_row(vr, kt, lane), acc[db]);
    }
  }

  // ---- normalize and store ctx as (b, s, e) bf16 --------------------------
  #pragma unroll
  for (int db = 0; db < 4; ++db) {
    #pragma unroll
    for (int v = 0; v < 8; ++v) {
      const int m = v + 8 * half;
      const float val = acc[db][v] / rs[v];
      ctx[((long)b * S_ + q0 + m) * E_ + h * HD_ + db * 16 + lc] = f2bf(val);
    }
  }
}

// ---------------------------------------------------------------------------
// Host: kernel_launch
// inputs: 0:x 1:Wq 2:bq 3:Wk 4:bk 5:Wv 6:bv 7:Wo 8:bo 9:Wp 10:rel_table
// ---------------------------------------------------------------------------
extern "C" void kernel_launch(void* const* d_in, const int* in_sizes, int n_in,
                              void* d_out, int out_size, void* d_ws,
                              size_t ws_size, hipStream_t stream) {
  const float* x   = (const float*)d_in[0];
  const float* Wq  = (const float*)d_in[1];
  const float* bq  = (const float*)d_in[2];
  const float* Wk  = (const float*)d_in[3];
  const float* bk  = (const float*)d_in[4];
  const float* Wv  = (const float*)d_in[5];
  const float* bv  = (const float*)d_in[6];
  const float* Wo  = (const float*)d_in[7];
  const float* bo  = (const float*)d_in[8];
  const float* Wp  = (const float*)d_in[9];
  const float* rel = (const float*)d_in[10];

  char* ws = (char*)d_ws;
  size_t off = 0;
  auto carve = [&](size_t bytes) {
    void* p = ws + off;
    off += (bytes + 255) & ~size_t(255);
    return p;
  };
  __bf16* xbf    = (__bf16*)carve((size_t)M_ * E_ * 2);       // 8 MB
  __bf16* Wqbf   = (__bf16*)carve((size_t)E_ * E_ * 2);       // 2 MB
  __bf16* Wkbf   = (__bf16*)carve((size_t)E_ * E_ * 2);
  __bf16* Wvbf   = (__bf16*)carve((size_t)E_ * E_ * 2);
  __bf16* Wobf   = (__bf16*)carve((size_t)E_ * E_ * 2);
  __bf16* Qbf    = (__bf16*)carve((size_t)M_ * E_ * 2);       // (b,h,s,d), pre-scaled
  __bf16* Kbf    = (__bf16*)carve((size_t)M_ * E_ * 2);       // (b,h,s,d)
  __bf16* Vtbf   = (__bf16*)carve((size_t)M_ * E_ * 2);       // (b,h,d,s)
  __bf16* relkbf = (__bf16*)carve((size_t)PROWS * HD_ * 2);   // 256 KB
  __bf16* ctxbf  = (__bf16*)carve((size_t)M_ * E_ * 2);       // (b,s,e)
  (void)ws_size; (void)in_sizes; (void)n_in; (void)out_size;

  const int nX2 = (M_ * E_) / 2, nW2 = (E_ * E_) / 2;
  cvt_f32_bf16<<<(nX2 + 255) / 256, 256, 0, stream>>>(x, xbf, nX2);
  cvt_f32_bf16<<<(nW2 + 255) / 256, 256, 0, stream>>>(Wq, Wqbf, nW2);
  cvt_f32_bf16<<<(nW2 + 255) / 256, 256, 0, stream>>>(Wk, Wkbf, nW2);
  cvt_f32_bf16<<<(nW2 + 255) / 256, 256, 0, stream>>>(Wv, Wvbf, nW2);
  cvt_f32_bf16<<<(nW2 + 255) / 256, 256, 0, stream>>>(Wo, Wobf, nW2);
  relkey_kernel<<<(PROWS * HD_ + 255) / 256, 256, 0, stream>>>(rel, Wp, relkbf);

  const float scale = 0.125f;  // 1/sqrt(HD)
  dim3 gg(M_ / 128, E_ / 64);  // (32, 16), 128 threads = 4 waves
  gemm_wt<0><<<gg, 128, 0, stream>>>(xbf, Wqbf, bq, Qbf, scale);  // scaled Q
  gemm_wt<0><<<gg, 128, 0, stream>>>(xbf, Wkbf, bk, Kbf, 1.0f);
  gemm_wt<2><<<gg, 128, 0, stream>>>(xbf, Wvbf, bv, Vtbf, 1.0f);

  attn_kernel<<<dim3(S_ / 16, B_ * H_), 32, 0, stream>>>(Qbf, Kbf, Vtbf,
                                                         relkbf, ctxbf);

  gemm_wt<3><<<gg, 128, 0, stream>>>(ctxbf, Wobf, bo, d_out, 1.0f);
}
